// AdaptiveComputationMLP_15659450761314
// MI455X (gfx1250) — matrix operand order repaired and checked by reading, not compiled, not run on hardware
//
#include <hip/hip_runtime.h>
#include <hip/hip_bf16.h>

typedef __attribute__((ext_vector_type(16))) __bf16 v16bf;
typedef __attribute__((ext_vector_type(8)))  __bf16 v8bf;
typedef __attribute__((ext_vector_type(8)))  float  v8f;
typedef __attribute__((ext_vector_type(4)))  int    v4i;

#define HIDDEN 1024
#define BLOCKD 256
#define NEXP   8
#define MTILE  32

#if __has_builtin(__builtin_amdgcn_global_load_async_to_lds_b128)
#define HAS_ASYNC_LDS 1
typedef __attribute__((address_space(1))) v4i*  gptr128;
typedef __attribute__((address_space(3))) v4i*  lptr128;
typedef __attribute__((address_space(1))) char* gcharp;
typedef __attribute__((address_space(3))) char* lcharp;
#else
#define HAS_ASYNC_LDS 0
#endif

// ---- helpers -------------------------------------------------------------

// 16-bit A-fragment (16x32): this lane's row, two 8-element K runs.
__device__ inline v16bf ld_frag_split8(const __bf16* p0, const __bf16* p1) {
    v8bf a = *(const v8bf*)p0;
    v8bf b = *(const v8bf*)p1;
    v16bf r;
#pragma unroll
    for (int i = 0; i < 8; ++i) { r[i] = a[i]; r[i + 8] = b[i]; }
    return r;
}

// 16-bit B-fragment (32x16): this lane's column, 16 contiguous K values.
__device__ inline v16bf ld_frag_cont16(const __bf16* p) {
    v8bf a = *(const v8bf*)p;
    v8bf b = *(const v8bf*)(p + 8);
    v16bf r;
#pragma unroll
    for (int i = 0; i < 8; ++i) { r[i] = a[i]; r[i + 8] = b[i]; }
    return r;
}

// Branchless gelu (tanh form); v_tanh_f32 when available on gfx1250.
__device__ inline float gelu_fast(float x) {
    float u = 0.7978845608028654f * (x + 0.044715f * x * x * x);
#if __has_builtin(__builtin_amdgcn_tanhf)
    float t = __builtin_amdgcn_tanhf(u);
#else
    float e = __expf(-2.0f * __builtin_fabsf(u));
    float t = __builtin_copysignf((1.0f - e) / (1.0f + e), u);
#endif
    return 0.5f * x * (1.0f + t);
}

// ---- kernel 1: highest = max(gating) ------------------------------------
__global__ __launch_bounds__(256) void acmlp_gmax(const int* __restrict__ g, int n,
                                                  int* __restrict__ outmax) {
    __shared__ int sm[256];
    int m = 0;
    for (int i = threadIdx.x; i < n; i += 256) m = max(m, g[i]);
    sm[threadIdx.x] = m;
    __syncthreads();
#pragma unroll
    for (int s = 128; s > 0; s >>= 1) {
        if (threadIdx.x < s) sm[threadIdx.x] = max(sm[threadIdx.x], sm[threadIdx.x + s]);
        __syncthreads();
    }
    if (threadIdx.x == 0) *outmax = sm[0];
}

// ---- kernel 2: f32 -> bf16 conversions -----------------------------------
// w1 [1024][2048] -> w1t [2048][1024] (transposed, column-contiguous B)
// w2 [2048][1024] -> w2t [1024][2048] (transposed)
// x  [N][1024]    -> xbf [N][1024]    (row-major, async-stageable)
__global__ __launch_bounds__(256) void acmlp_cvt(const float* __restrict__ w1,
                                                 const float* __restrict__ w2,
                                                 const float* __restrict__ x,
                                                 __bf16* __restrict__ w1t,
                                                 __bf16* __restrict__ w2t,
                                                 __bf16* __restrict__ xbf, long nx) {
    const long stride = (long)gridDim.x * blockDim.x;
    const long tid0   = (long)blockIdx.x * blockDim.x + threadIdx.x;

    const long T1 = (long)HIDDEN * (NEXP * BLOCKD);
    for (long i = tid0; i < T1; i += stride) {
        long r = i / (NEXP * BLOCKD), c = i % (NEXP * BLOCKD);
        w1t[c * HIDDEN + r] = (__bf16)w1[i];
    }
    const long T2 = (long)(NEXP * BLOCKD) * HIDDEN;
    for (long i = tid0; i < T2; i += stride) {
        long r = i / HIDDEN, c = i % HIDDEN;
        w2t[c * (NEXP * BLOCKD) + r] = (__bf16)w2[i];
    }
    const float4* xs4 = (const float4*)x;
    for (long i = tid0; i < nx / 4; i += stride) {
        float4 v = xs4[i];
        __bf16* d = xbf + i * 4;
        d[0] = (__bf16)v.x; d[1] = (__bf16)v.y;
        d[2] = (__bf16)v.z; d[3] = (__bf16)v.w;
    }
}

// ---- kernel 3: fused adaptive MLP ----------------------------------------
// Grid: N/32 workgroups of 256 threads (8 wave32). Each WG owns 32 rows.
// Per expert b: H = gelu(mask * (X @ W1_b)) in LDS, acc += H @ W2_b.
__global__ __launch_bounds__(256) void acmlp_main(const __bf16* __restrict__ xbf,
                                                  const int* __restrict__ gating,
                                                  const __bf16* __restrict__ w1t,
                                                  const __bf16* __restrict__ w2t,
                                                  const int* __restrict__ hptr,
                                                  float* __restrict__ out) {
    __shared__ __align__(16) __bf16 xs[MTILE * HIDDEN];  // 64 KB X tile (bf16)
    __shared__ __align__(16) __bf16 hs[MTILE * BLOCKD];  // 16 KB H tile (bf16)
    __shared__ int gs[MTILE];

    const int tid    = threadIdx.x;
    const int wave   = tid >> 5;
    const int lane   = tid & 31;
    const int lane16 = lane & 15;
    const int hihalf = lane >> 4;        // 0: lanes 0-15, 1: lanes 16-31
    const int row0   = blockIdx.x * MTILE;

    // --- stage bf16 X tile (64 KB contiguous) into LDS ---
#if HAS_ASYNC_LDS
    {
        gcharp gbase = (gcharp)(const char*)(xbf + (long)row0 * HIDDEN);
        lcharp lbase = (lcharp)(char*)xs;
#pragma unroll
        for (int j = 0; j < 16; ++j) {
            int chunk = j * 256 + tid;   // 16B chunks, 4096 total
            __builtin_amdgcn_global_load_async_to_lds_b128(
                (gptr128)(gbase + chunk * 16), (lptr128)(lbase + chunk * 16), 0, 0);
        }
#if __has_builtin(__builtin_amdgcn_s_wait_asynccnt)
        __builtin_amdgcn_s_wait_asynccnt(0);
#else
        asm volatile("s_wait_asynccnt 0x0" ::: "memory");
#endif
    }
#else
    {
        const v8bf* src = (const v8bf*)(xbf + (long)row0 * HIDDEN);
        v8bf* dst = (v8bf*)xs;
#pragma unroll
        for (int j = 0; j < 16; ++j) {
            int chunk = j * 256 + tid;
            dst[chunk] = src[chunk];
        }
    }
#endif
    if (tid < MTILE) gs[tid] = gating[row0 + tid];
    __syncthreads();

    const int highest = *hptr;
    const int koff    = hihalf * 8;              // A-frag K sub-offset
    const int kboff   = hihalf * 16;             // unused; B frags use koff

    v8f acc[16];                                 // [m][s]: 2 row tiles x 8 col tiles
#pragma unroll
    for (int s = 0; s < 16; ++s) acc[s] = (v8f){};

#pragma unroll 1
    for (int b = 0; b < NEXP; ++b) {
        // ===== GEMM1: wave computes H[32 rows x 32 cols at wave*32] =====
        v8f c00 = (v8f){}, c01 = (v8f){}, c10 = (v8f){}, c11 = (v8f){};
        const __bf16* w1c0 = w1t + (long)(b * BLOCKD + wave * 32 + lane16) * HIDDEN;
        const __bf16* w1c1 = w1c0 + 16L * HIDDEN;
#pragma unroll 4
        for (int k = 0; k < HIDDEN; k += 32) {
            v16bf a0 = ld_frag_split8(&xs[lane16 * HIDDEN + k + koff],
                                      &xs[lane16 * HIDDEN + k + koff + 16]);
            v16bf a1 = ld_frag_split8(&xs[(16 + lane16) * HIDDEN + k + koff],
                                      &xs[(16 + lane16) * HIDDEN + k + koff + 16]);
            v16bf b0 = ld_frag_split8(w1c0 + k + koff, w1c0 + k + koff + 16);
            v16bf b1 = ld_frag_split8(w1c1 + k + koff, w1c1 + k + koff + 16);
            c00 = __builtin_amdgcn_wmma_f32_16x16x32_bf16(false, a0, false, b0,
                                                          (short)0, c00, false, false);
            c01 = __builtin_amdgcn_wmma_f32_16x16x32_bf16(false, a0, false, b1,
                                                          (short)0, c01, false, false);
            c10 = __builtin_amdgcn_wmma_f32_16x16x32_bf16(false, a1, false, b0,
                                                          (short)0, c10, false, false);
            c11 = __builtin_amdgcn_wmma_f32_16x16x32_bf16(false, a1, false, b1,
                                                          (short)0, c11, false, false);
        }

        // ===== gelu + row mask, store H tile to LDS (bf16) =====
#pragma unroll
        for (int r = 0; r < 8; ++r) {
            int rr0 = r + hihalf * 8;            // D-layout row, row-subtile 0
            int rr1 = 16 + rr0;                  // row-subtile 1
            float m0 = (gs[rr0] >= b + 1 && (b + 1) < highest) ? 1.0f : 0.0f;
            float m1 = (gs[rr1] >= b + 1 && (b + 1) < highest) ? 1.0f : 0.0f;
            hs[rr0 * BLOCKD + wave * 32 + lane16]      = (__bf16)(gelu_fast(c00[r]) * m0);
            hs[rr0 * BLOCKD + wave * 32 + 16 + lane16] = (__bf16)(gelu_fast(c01[r]) * m0);
            hs[rr1 * BLOCKD + wave * 32 + lane16]      = (__bf16)(gelu_fast(c10[r]) * m1);
            hs[rr1 * BLOCKD + wave * 32 + 16 + lane16] = (__bf16)(gelu_fast(c11[r]) * m1);
        }
        __syncthreads();

        // ===== GEMM2: acc[32 x 128strip] += H[32x256] @ W2_b[256x128strip] =====
        const __bf16* w2base = w2t + (long)b * BLOCKD;
#pragma unroll 1
        for (int k = 0; k < BLOCKD; k += 32) {
            v16bf a0 = ld_frag_split8(&hs[lane16 * BLOCKD + k + koff],
                                      &hs[lane16 * BLOCKD + k + koff + 16]);
            v16bf a1 = ld_frag_split8(&hs[(16 + lane16) * BLOCKD + k + koff],
                                      &hs[(16 + lane16) * BLOCKD + k + koff + 16]);
#pragma unroll
            for (int s = 0; s < 8; ++s) {
                int col = wave * 128 + s * 16 + lane16;
                v16bf bf = ld_frag_split8(w2base + (long)col * (NEXP * BLOCKD) + k + koff, w2base + (long)col * (NEXP * BLOCKD) + k + koff + 16);
                acc[s]     = __builtin_amdgcn_wmma_f32_16x16x32_bf16(false, a0, false, bf,
                                                                     (short)0, acc[s], false, false);
                acc[8 + s] = __builtin_amdgcn_wmma_f32_16x16x32_bf16(false, a1, false, bf,
                                                                     (short)0, acc[8 + s], false, false);
            }
        }
        __syncthreads();                          // hs reused next expert
    }

    // ===== write out: each wave owns a 128-column strip of the 32 rows =====
#pragma unroll
    for (int m = 0; m < 2; ++m) {
#pragma unroll
        for (int s = 0; s < 8; ++s) {
#pragma unroll
            for (int r = 0; r < 8; ++r) {
                int rr  = m * 16 + r + hihalf * 8;
                int col = wave * 128 + s * 16 + lane16;
                out[(long)(row0 + rr) * HIDDEN + col] = acc[m * 8 + s][r];
            }
        }
    }
}

// ---- launcher ------------------------------------------------------------
extern "C" void kernel_launch(void* const* d_in, const int* in_sizes, int n_in,
                              void* d_out, int out_size, void* d_ws, size_t ws_size,
                              hipStream_t stream) {
    const float* x      = (const float*)d_in[0];
    const int*   gating = (const int*)d_in[1];
    const float* w1     = (const float*)d_in[2];
    const float* w2     = (const float*)d_in[3];
    float*       out    = (float*)d_out;

    char* ws = (char*)d_ws;
    int*    hptr = (int*)ws;
    __bf16* w1t  = (__bf16*)(ws + 128);
    __bf16* w2t  = (__bf16*)(ws + 128 + (size_t)(NEXP * BLOCKD) * HIDDEN * 2);
    __bf16* xbf  = (__bf16*)(ws + 128 + (size_t)(NEXP * BLOCKD) * HIDDEN * 4);

    const long nx = (long)in_sizes[0];            // N * HIDDEN
    const int  N  = (int)(nx / HIDDEN);           // 32768

    acmlp_gmax<<<1, 256, 0, stream>>>(gating, N, hptr);
    acmlp_cvt<<<4096, 256, 0, stream>>>(w1, w2, x, w1t, w2t, xbf, nx);
    acmlp_main<<<N / MTILE, 256, 0, stream>>>(xbf, gating, w1t, w2t, hptr, out);
}
